// MultiInputGAT_6399501271418
// MI455X (gfx1250) — matrix operand order repaired and checked by reading, not compiled, not run on hardware
//
#include <hip/hip_runtime.h>
#include <hip/hip_bf16.h>

// ---------------------------------------------------------------------------
// MI455X (gfx1250) implementation of the (dead-code-eliminated) MultiInputGAT.
// Only stream 1 is live; GEMMs run on v_wmma_f32_16x16x32_bf16 (wave32),
// with fragment-native pre-packed weights (b128 loads) and 16x64 wave tiles.
// ---------------------------------------------------------------------------

typedef __bf16          v16bf __attribute__((ext_vector_type(16)));
typedef float           v8f   __attribute__((ext_vector_type(8)));
typedef unsigned short  us8   __attribute__((ext_vector_type(8)));

#define DIN  128
#define DH   512
#define BN_EPS 1e-5f

__device__ __forceinline__ unsigned short f2bf(float f) {
    unsigned u = __float_as_uint(f);
    unsigned r = u + 0x7fffu + ((u >> 16) & 1u);   // round-to-nearest-even
    return (unsigned short)(r >> 16);
}
__device__ __forceinline__ unsigned fkey(float f) {   // order-preserving uint key
    unsigned u = __float_as_uint(f);
    return (u & 0x80000000u) ? ~u : (u | 0x80000000u);
}
__device__ __forceinline__ float kdec(unsigned k) {
    unsigned u = (k & 0x80000000u) ? (k & 0x7fffffffu) : ~k;
    return __uint_as_float(u);
}

// ---------------- elementwise helpers ----------------

__global__ void k_cvt_bf16(const float* __restrict__ in, unsigned short* __restrict__ out, int n) {
    int i = blockIdx.x * blockDim.x + threadIdx.x;
    if (i < n) out[i] = f2bf(in[i]);
}

__global__ void k_zero_f32(float* __restrict__ p, int n) {
    int i = blockIdx.x * blockDim.x + threadIdx.x;
    if (i < n) p[i] = 0.0f;
}

__global__ void k_init_nodes(unsigned* __restrict__ menc, float* __restrict__ den, int n) {
    int i = blockIdx.x * blockDim.x + threadIdx.x;
    if (i < n) { menc[i] = 0u; den[i] = 0.0f; }   // 0u == below key(-inf)
}

__global__ void k_bias_relu(float* __restrict__ h, const float* __restrict__ bias, int n, int C) {
    int i = blockIdx.x * blockDim.x + threadIdx.x;
    if (i < n) h[i] = fmaxf(h[i] + bias[i % C], 0.0f);
}

// ---------------- weight packing: row-major f32 [K][Nc] -> WMMA B-fragment order ----
// Bp[ ((nt*ksteps + kb)*32 + lane)*16 + j ] = bf16( B[kb*32 + kbase(lane) + j + 8*(j>>3)][nt*16 + (lane&15)] )
// so each lane's 16-element fragment is 32 contiguous bytes (2 x b128 loads).

__global__ void k_pack_w(const float* __restrict__ B, unsigned short* __restrict__ Bp,
                         int K, int Nc) {
    int i = blockIdx.x * blockDim.x + threadIdx.x;
    if (i >= K * Nc) return;
    const int ksteps = K >> 5;
    const int j    = i & 15;
    const int lane = (i >> 4) & 31;
    const int g    = i >> 9;            // nt*ksteps + kb
    const int kb   = g % ksteps;
    const int nt   = g / ksteps;
    const int kbase = (lane < 16) ? 0 : 8;     // B frag: lanes 16-31 carry K = 8..15 / 24..31
    const int k = kb * 32 + kbase + ((j < 8) ? j : (8 + j));
    const int n = nt * 16 + (lane & 15);
    Bp[i] = f2bf(B[(size_t)k * Nc + n]);
}

// ---------------- WMMA bf16 GEMM:  C[M,Nc] = A[M,K] * Bp + bias (opt relu) ----------
// One wave per 16 x (16*NTW) output strip; A frag shared across NTW WMMAs per K-step.

template <int NTW>
__global__ void k_wmma_gemm_bf16(const unsigned short* __restrict__ A,
                                 const unsigned short* __restrict__ Bp,
                                 const float* __restrict__ bias,
                                 float* __restrict__ C,
                                 int M, int K, int Nc, int relu) {
    const int wave  = (blockIdx.x * blockDim.x + threadIdx.x) >> 5;
    const int lane  = threadIdx.x & 31;
    const int Ng    = (Nc >> 4) / NTW;
    const int tiles = (M >> 4) * Ng;
    if (wave >= tiles) return;                 // whole wave retires: EXEC stays all-ones
    const int mt  = wave / Ng;
    const int nt0 = (wave - mt * Ng) * NTW;

    const int half   = lane >> 4;
    const int l15    = lane & 15;
    const int akbase = half ? 8 : 0;           // A: lanes 16-31 carry K = 8..15 / 24..31
    const unsigned short* Arow = A + (size_t)(mt * 16 + l15) * K;
    const int ksteps = K >> 5;

    union F16 { us8 v[2]; v16bf bf; };

    v8f acc[NTW];
#pragma unroll
    for (int t = 0; t < NTW; ++t) acc[t] = (v8f){0.f, 0.f, 0.f, 0.f, 0.f, 0.f, 0.f, 0.f};

#pragma unroll 4
    for (int kb = 0; kb < ksteps; ++kb) {
        if (kb + 1 < ksteps)                   // -> global_prefetch_b8
            __builtin_prefetch(Arow + (kb + 1) * 32, 0, 3);

        F16 ua;                                // A frag: 2 contiguous 16B runs
        ua.v[0] = *(const us8*)(Arow + kb * 32 + akbase);
        ua.v[1] = *(const us8*)(Arow + kb * 32 + akbase + 16);

#pragma unroll
        for (int t = 0; t < NTW; ++t) {        // B frags: fragment-native, contiguous
            F16 ub;
            const unsigned short* bp =
                Bp + ((size_t)((nt0 + t) * ksteps + kb) * 32 + lane) * 16;
            ub.v[0] = *(const us8*)bp;
            ub.v[1] = *(const us8*)(bp + 8);
            acc[t] = __builtin_amdgcn_wmma_f32_16x16x32_bf16(
                         false, ua.bf, false, ub.bf, (short)0, acc[t], false, false);
        }
    }

    const int mofs = half ? 8 : 0;             // C/D: VGPR r -> M = r (+8 for lanes 16-31)
#pragma unroll
    for (int t = 0; t < NTW; ++t) {
        const int ncol = (nt0 + t) * 16 + l15;
        const float bv = bias ? bias[ncol] : 0.0f;
#pragma unroll
        for (int r = 0; r < 8; ++r) {
            float v = acc[t][r] + bv;
            if (relu) v = fmaxf(v, 0.0f);
            C[(size_t)(mt * 16 + r + mofs) * Nc + ncol] = v;
        }
    }
}

// ---------------- GATv2 edge kernels (one wave per edge, 512 = 32 lanes x 16) --------

__global__ void k_edge_logits(const int* __restrict__ ei,
                              const float* __restrict__ xl, const float* __restrict__ xr,
                              const float* __restrict__ att,
                              float* __restrict__ logits, unsigned* __restrict__ menc,
                              int E, int Nn) {
    const int wid  = (blockIdx.x * blockDim.x + threadIdx.x) >> 5;
    const int lane = threadIdx.x & 31;
    if (wid >= E + Nn) return;
    int src, dst;
    if (wid < E) { src = ei[wid]; dst = ei[E + wid]; }
    else         { src = dst = wid - E; }      // self loop

    const float4* pl = (const float4*)(xl  + (size_t)src * DH + lane * 16);
    const float4* pr = (const float4*)(xr  + (size_t)dst * DH + lane * 16);
    const float4* pa = (const float4*)(att + lane * 16);
    float acc = 0.0f;
#pragma unroll
    for (int q = 0; q < 4; ++q) {
        float4 a = pl[q], b = pr[q], w = pa[q];
        float s0 = a.x + b.x, s1 = a.y + b.y, s2 = a.z + b.z, s3 = a.w + b.w;
        s0 = (s0 > 0.f) ? s0 : 0.2f * s0;      // leaky_relu(., 0.2)
        s1 = (s1 > 0.f) ? s1 : 0.2f * s1;
        s2 = (s2 > 0.f) ? s2 : 0.2f * s2;
        s3 = (s3 > 0.f) ? s3 : 0.2f * s3;
        acc += s0 * w.x + s1 * w.y + s2 * w.z + s3 * w.w;
    }
#pragma unroll
    for (int o = 16; o > 0; o >>= 1) acc += __shfl_xor(acc, o, 32);
    if (lane == 0) {
        logits[wid] = acc;
        atomicMax(&menc[dst], fkey(acc));      // segment_max
    }
}

__global__ void k_edge_exp(const int* __restrict__ ei,
                           const float* __restrict__ logits, const unsigned* __restrict__ menc,
                           float* __restrict__ ex, float* __restrict__ den,
                           int E, int Nn) {
    int e = blockIdx.x * blockDim.x + threadIdx.x;
    if (e >= E + Nn) return;
    int dst = (e < E) ? ei[E + e] : (e - E);
    float v = __expf(logits[e] - kdec(menc[dst]));
    ex[e] = v;
    atomicAdd(&den[dst], v);                   // segment_sum
}

__global__ void k_edge_agg(const int* __restrict__ ei,
                           const float* __restrict__ xl,
                           const float* __restrict__ ex, const float* __restrict__ den,
                           float* __restrict__ h, int E, int Nn) {
    const int wid  = (blockIdx.x * blockDim.x + threadIdx.x) >> 5;
    const int lane = threadIdx.x & 31;
    if (wid >= E + Nn) return;
    int src, dst;
    if (wid < E) { src = ei[wid]; dst = ei[E + wid]; }
    else         { src = dst = wid - E; }
    const float alpha = ex[wid] / den[dst];
    const float4* pl = (const float4*)(xl + (size_t)src * DH + lane * 16);
    float* ph = h + (size_t)dst * DH + lane * 16;
#pragma unroll
    for (int q = 0; q < 4; ++q) {
        float4 a = pl[q];
        atomicAdd(ph + 4 * q + 0, a.x * alpha);
        atomicAdd(ph + 4 * q + 1, a.y * alpha);
        atomicAdd(ph + 4 * q + 2, a.z * alpha);
        atomicAdd(ph + 4 * q + 3, a.w * alpha);
    }
}

// ---------------- BatchNorm (training-mode batch stats, biased var) ----------------

__global__ void k_bn_stats(const float* __restrict__ X, float* __restrict__ stats,
                           int rows, int C) {
    const int c = blockIdx.x;
    float s = 0.f, s2 = 0.f;
    for (int r = threadIdx.x; r < rows; r += blockDim.x) {
        float v = X[(size_t)r * C + c];
        s += v; s2 += v * v;
    }
    __shared__ float sh[256], sh2[256];
    sh[threadIdx.x] = s; sh2[threadIdx.x] = s2;
    __syncthreads();
    for (int st = 128; st > 0; st >>= 1) {
        if (threadIdx.x < st) { sh[threadIdx.x] += sh[threadIdx.x + st];
                                sh2[threadIdx.x] += sh2[threadIdx.x + st]; }
        __syncthreads();
    }
    if (threadIdx.x == 0) {
        float mu = sh[0] / rows;
        stats[c]     = mu;
        stats[C + c] = sh2[0] / rows - mu * mu;   // biased variance
    }
}

__global__ void k_bn_apply_bf16(const float* __restrict__ X, const float* __restrict__ stats,
                                const float* __restrict__ g, const float* __restrict__ b,
                                unsigned short* __restrict__ out, int rows, int C) {
    size_t i = (size_t)blockIdx.x * blockDim.x + threadIdx.x;
    if (i >= (size_t)rows * C) return;
    int c = (int)(i % C);
    float v = g[c] * (X[i] - stats[c]) * rsqrtf(stats[C + c] + BN_EPS) + b[c];
    out[i] = f2bf(v);
}

// ---------------- head: sigmoid + log_softmax over 16 cols ----------------

__global__ void k_head(const float* __restrict__ Z, float* __restrict__ out, int rows) {
    int r = blockIdx.x * blockDim.x + threadIdx.x;
    if (r >= rows) return;
    float v[16], mx = -INFINITY;
#pragma unroll
    for (int j = 0; j < 16; ++j) { v[j] = Z[(size_t)r * 16 + j]; mx = fmaxf(mx, v[j]); }
    float s = 0.f;
#pragma unroll
    for (int j = 0; j < 16; ++j) s += __expf(v[j] - mx);
    float lse = __logf(s);
#pragma unroll
    for (int j = 0; j < 16; ++j)
        out[(size_t)r * 16 + j] = 1.0f / (1.0f + __expf(-v[j]));              // sigmoid
#pragma unroll
    for (int j = 0; j < 16; ++j)
        out[(size_t)rows * 16 + (size_t)r * 16 + j] = v[j] - mx - lse;        // log_softmax
}

// ---------------------------------------------------------------------------

extern "C" void kernel_launch(void* const* d_in, const int* in_sizes, int n_in,
                              void* d_out, int out_size, void* d_ws, size_t ws_size,
                              hipStream_t stream) {
    (void)n_in; (void)out_size; (void)ws_size;
    // Stream 1 (stream 0 is dead code: h is overwritten before use).
    const float* data1 = (const float*)d_in[11];
    const int*   ei    = (const int*)  d_in[12];
    const float* Wl    = (const float*)d_in[14];
    const float* bl    = (const float*)d_in[15];
    const float* Wr    = (const float*)d_in[16];
    const float* br    = (const float*)d_in[17];
    const float* att   = (const float*)d_in[18];
    const float* bias  = (const float*)d_in[19];
    const float* gamma = (const float*)d_in[20];
    const float* beta  = (const float*)d_in[21];
    const float* W1 = (const float*)d_in[22]; const float* b1  = (const float*)d_in[23];
    const float* g1 = (const float*)d_in[24]; const float* be1 = (const float*)d_in[25];
    const float* W2 = (const float*)d_in[26]; const float* b2  = (const float*)d_in[27];
    const float* g2 = (const float*)d_in[28]; const float* be2 = (const float*)d_in[29];
    const float* W3 = (const float*)d_in[30]; const float* b3  = (const float*)d_in[31];

    const int N    = in_sizes[11] / DIN;   // 10000
    const int E    = in_sizes[12] / 2;     // 160000
    const int Etot = E + N;

    // ---- workspace layout (aggressive reuse, ~74 MB) ----
    char* ws = (char*)d_ws;
    size_t off = 0;
    auto take = [&](size_t bytes) -> char* {
        char* p = ws + off;
        off = (off + bytes + 255) & ~(size_t)255;
        return p;
    };
    float* xl = (float*)take((size_t)N * DH * 4);
    char*  regB = take((size_t)N * DH * 4);            // xr, later z1/z2/z2bf/z3
    float*          xr   = (float*)regB;
    float*          z1   = (float*)regB;
    float*          z2   = (float*)(regB + (size_t)N * 256 * 4);
    unsigned short* z2bf = (unsigned short*)(regB + (size_t)N * 256 * 4 + (size_t)N * 128 * 4);
    float*          z3   = (float*)(regB + (size_t)N * 256 * 4 + (size_t)N * 128 * 4 + (size_t)N * 128 * 2);
    char*  regH = take((size_t)N * DH * 4);            // h, later z1bf
    float*          h    = (float*)regH;
    unsigned short* z1bf = (unsigned short*)regH;
    unsigned short* regA = (unsigned short*)take((size_t)N * DH * 2);  // xbf, later hbf
    unsigned short* xbf  = regA;
    unsigned short* hbf  = regA;
    unsigned short* Wlp = (unsigned short*)take((size_t)DIN * DH * 2);
    unsigned short* Wrp = (unsigned short*)take((size_t)DIN * DH * 2);
    unsigned short* W1p = (unsigned short*)take((size_t)512 * 256 * 2);
    unsigned short* W2p = (unsigned short*)take((size_t)256 * 128 * 2);
    unsigned short* W3p = (unsigned short*)take((size_t)128 * 16 * 2);
    float*    logits = (float*)   take((size_t)Etot * 4);
    float*    ex     = (float*)   take((size_t)Etot * 4);
    unsigned* menc   = (unsigned*)take((size_t)N * 4);
    float*    den    = (float*)   take((size_t)N * 4);
    float*    stats  = (float*)   take((size_t)2 * DH * 4);

    const int T = 256;
    auto cdiv = [](long long a, long long b) -> int { return (int)((a + b - 1) / b); };
    auto gemm = [&](const unsigned short* A, const unsigned short* Bp, const float* bi,
                    float* C, int M, int K, int Nc, int relu) {
        if (Nc % 64 == 0) {
            int tiles = (M / 16) * (Nc / 64);
            k_wmma_gemm_bf16<4><<<cdiv((long long)tiles * 32, T), T, 0, stream>>>(A, Bp, bi, C, M, K, Nc, relu);
        } else {
            int tiles = (M / 16) * (Nc / 16);
            k_wmma_gemm_bf16<1><<<cdiv((long long)tiles * 32, T), T, 0, stream>>>(A, Bp, bi, C, M, K, Nc, relu);
        }
    };

    // activations -> bf16; weights -> fragment-native packed bf16
    k_cvt_bf16<<<cdiv((long long)N * DIN, T), T, 0, stream>>>(data1, xbf, N * DIN);
    k_pack_w<<<cdiv(DIN * DH, T), T, 0, stream>>>(Wl, Wlp, DIN, DH);
    k_pack_w<<<cdiv(DIN * DH, T), T, 0, stream>>>(Wr, Wrp, DIN, DH);
    k_pack_w<<<cdiv(512 * 256, T), T, 0, stream>>>(W1, W1p, 512, 256);
    k_pack_w<<<cdiv(256 * 128, T), T, 0, stream>>>(W2, W2p, 256, 128);
    k_pack_w<<<cdiv(128 * 16, T), T, 0, stream>>>(W3, W3p, 128, 16);

    // xl = x@Wl + bl ; xr = x@Wr + br   (WMMA bf16, f32 accumulate)
    gemm(xbf, Wlp, bl, xl, N, DIN, DH, 0);
    gemm(xbf, Wrp, br, xr, N, DIN, DH, 0);

    // GATv2 edge softmax + aggregation
    k_zero_f32<<<cdiv((long long)N * DH, T), T, 0, stream>>>(h, N * DH);
    k_init_nodes<<<cdiv(N, T), T, 0, stream>>>(menc, den, N);
    k_edge_logits<<<cdiv((long long)Etot * 32, T), T, 0, stream>>>(ei, xl, xr, att, logits, menc, E, N);
    k_edge_exp<<<cdiv(Etot, T), T, 0, stream>>>(ei, logits, menc, ex, den, E, N);
    k_edge_agg<<<cdiv((long long)Etot * 32, T), T, 0, stream>>>(ei, xl, ex, den, h, E, N);
    k_bias_relu<<<cdiv((long long)N * DH, T), T, 0, stream>>>(h, bias, N * DH, DH);

    // BN(stream1) -> bf16
    k_bn_stats<<<DH, T, 0, stream>>>(h, stats, N, DH);
    k_bn_apply_bf16<<<cdiv((long long)N * DH, T), T, 0, stream>>>(h, stats, gamma, beta, hbf, N, DH);

    // MLP head: 512 -> 256 -> 128 -> 16
    gemm(hbf, W1p, b1, z1, N, 512, 256, 1);
    k_bn_stats<<<256, T, 0, stream>>>(z1, stats, N, 256);
    k_bn_apply_bf16<<<cdiv((long long)N * 256, T), T, 0, stream>>>(z1, stats, g1, be1, z1bf, N, 256);

    gemm(z1bf, W2p, b2, z2, N, 256, 128, 1);
    k_bn_stats<<<128, T, 0, stream>>>(z2, stats, N, 128);
    k_bn_apply_bf16<<<cdiv((long long)N * 128, T), T, 0, stream>>>(z2, stats, g2, be2, z2bf, N, 128);

    gemm(z2bf, W3p, b3, z3, N, 128, 16, 0);

    // sigmoid + log_softmax -> d_out (concatenated)
    k_head<<<cdiv(N, T), T, 0, stream>>>(z3, (float*)d_out, N);
}
